// BiLstmCrf_41686952574981
// MI455X (gfx1250) — matrix operand order repaired and checked by reading, not compiled, not run on hardware
//
#include <hip/hip_runtime.h>
#include <hip/hip_bf16.h>
#include <stdint.h>

// ---------------- CDNA5 WMMA types ----------------
typedef __attribute__((ext_vector_type(16))) __bf16        v16bf;
typedef __attribute__((ext_vector_type(8)))  float         v8f;
typedef __attribute__((ext_vector_type(4)))  float         v4f;
typedef __attribute__((ext_vector_type(4)))  unsigned int  v4u;
typedef __attribute__((ext_vector_type(4)))  int           v4i;

#define TT    512
#define BB    128
#define EE    300
#define HHh   256
#define GN    1024          /* 4*Hh */
#define KT    20
#define MM    (TT*BB)       /* 65536 */
#define KPAD0 320
#define PRE_T_STRIDE (128*4*32*8)   /* floats per timestep in permuted pre layout */

// ---- optional CDNA5 async memory->LDS path (compile-safe feature probe) ----
#if defined(__has_builtin)
# if __has_builtin(__builtin_amdgcn_global_load_async_to_lds_b128) && \
     __has_builtin(__builtin_amdgcn_s_wait_asynccnt)
#  define USE_ASYNC_LDS 1
# endif
#endif
#ifndef USE_ASYNC_LDS
# define USE_ASYNC_LDS 0
#endif
// builtin signature (from hipcc diagnostic): (v4i AS1*, v4i AS3*, imm offset, imm cpol)
#define GPTR(p) ((__attribute__((address_space(1))) v4i*)(p))
#define LPTR(p) ((__attribute__((address_space(3))) v4i*)(p))

__device__ __forceinline__ unsigned short f2bf(float f) {
  unsigned int u = __float_as_uint(f);
  u += 0x7FFFu + ((u >> 16) & 1u);     // round-to-nearest-even
  return (unsigned short)(u >> 16);
}
__device__ __forceinline__ float sig_(float x) { return 1.0f / (1.0f + __expf(-x)); }
__device__ __forceinline__ float tanh_(float x) {
  x = fminf(fmaxf(x, -15.0f), 15.0f);
  float e = __expf(2.0f * x);
  return (e - 1.0f) / (e + 1.0f);
}

union ABf { v16bf v; v4u q[2]; unsigned short u[16]; };
union F8  { v8f v; v4f h[2]; };

// A fragment (16x32 bf16): lane holds two contiguous 16B chunks of its row:
// K in [8*lh, 8*lh+7] and [16+8*lh, 16+8*lh+7].  p = rowBase + 8*lh.
__device__ __forceinline__ v16bf load_a_frag_v(const unsigned short* p) {
  ABf r;
  r.q[0] = *(const v4u*)(p);
  r.q[1] = *(const v4u*)(p + 16);
  return r.v;
}
// B fragment (32x16 bf16, element (k,n) at colBase[n*ldb + k]): lane holds two
// contiguous 16B chunks K in [8*lh, 8*lh+7] and [16+8*lh, 16+8*lh+7].  p = colBase + n*ldb + 8*lh.
__device__ __forceinline__ v16bf load_b_frag_v(const unsigned short* p) {
  ABf r;
  r.q[0] = *(const v4u*)(p);
  r.q[1] = *(const v4u*)(p + 16);
  return r.v;
}

// ---------------- small conversion / setup kernels ----------------
__global__ void cvt_pad_bf16_kernel(const float* __restrict__ src,
                                    unsigned short* __restrict__ dst,
                                    int rows, int cols, int colsPad) {
  int idx = blockIdx.x * blockDim.x + threadIdx.x;
  if (idx >= rows * colsPad) return;
  int r = idx / colsPad, c = idx - r * colsPad;
  dst[idx] = (c < cols) ? f2bf(src[(size_t)r * cols + c]) : (unsigned short)0;
}

__global__ void add_bias_kernel(const float* a, const float* b, float* c, int n) {
  int i = blockIdx.x * blockDim.x + threadIdx.x;
  if (i < n) c[i] = a[i] + b[i];
}

__global__ void embed_kernel(const int* __restrict__ x,
                             const float* __restrict__ emb,
                             unsigned short* __restrict__ xin) {
  int idx = blockIdx.x * blockDim.x + threadIdx.x;     // MM * KPAD0
  if (idx >= MM * KPAD0) return;
  int row = idx / KPAD0, c = idx - row * KPAD0;
  int tok = x[row];
  xin[idx] = (c < EE) ? f2bf(emb[(size_t)tok * EE + c]) : (unsigned short)0;
}

// ---------------- input GEMM ----------------
// pre_p[t][tile][gate][lane][e] permuted output so the scan can seed its WMMA
// accumulators with contiguous b128 loads.  blockIdx.y == timestep t.
// Double-buffered LDS staging; async memory->LDS copies when available.
__global__ __launch_bounds__(256)
void gemm_bias_kernel(const unsigned short* __restrict__ A,
                      const unsigned short* __restrict__ Bw,
                      const float* __restrict__ bias,
                      float* __restrict__ prep,
                      int Kp) {
  __shared__ unsigned short As[2][128 * 32];   // 2 x 8 KB
  __shared__ unsigned short Bs[2][64 * 32];    // 2 x 4 KB
  int tid  = threadIdx.x;
  int wave = tid >> 5, lane = tid & 31;
  int lh = lane >> 4, nlm = lane & 15;
  int n0 = blockIdx.x * 64;
  int t  = blockIdx.y;                      // m0 = t*128, full batch per block
  int wm = wave >> 1, wn = wave & 1;        // 4x2 wave grid
  int mW = wm * 32, nW = wn * 32;
  v8f acc[2][2] = {};

  // staging assignment (b128 granules)
  int arow = tid >> 1, acolU = (tid & 1) * 16;   // A: 128 rows x 32 ushorts, 32B/thread
  int brow = tid >> 2, bcolU = (tid & 3) * 8;    // B: 64 rows x 32 ushorts, 16B/thread
  const unsigned short* Arow = A + (size_t)(t * 128 + arow) * Kp + acolU;
  const unsigned short* Brow = Bw + (size_t)(n0 + brow) * Kp + bcolU;
  unsigned short* lA0 = &As[0][arow * 32 + acolU];
  unsigned short* lA1 = &As[1][arow * 32 + acolU];
  unsigned short* lB0 = &Bs[0][brow * 32 + bcolU];
  unsigned short* lB1 = &Bs[1][brow * 32 + bcolU];

  const int NK = Kp >> 5;
#if USE_ASYNC_LDS
  // prologue: stage k-step 0 into buffer 0 directly in LDS
  __builtin_amdgcn_global_load_async_to_lds_b128(GPTR(Arow), LPTR(lA0), 0, 0);
  __builtin_amdgcn_global_load_async_to_lds_b128(GPTR(Arow + 8), LPTR(lA0 + 8), 0, 0);
  __builtin_amdgcn_global_load_async_to_lds_b128(GPTR(Brow), LPTR(lB0), 0, 0);
  __builtin_amdgcn_s_wait_asynccnt(0);
  __syncthreads();
#else
  {
    v4u a0 = *(const v4u*)(Arow);
    v4u a1 = *(const v4u*)(Arow + 8);
    v4u b0 = *(const v4u*)(Brow);
    *(v4u*)(lA0) = a0;
    *(v4u*)(lA0 + 8) = a1;
    *(v4u*)(lB0) = b0;
  }
  __syncthreads();
#endif

  for (int kk = 0; kk < NK; ++kk) {
    int cur = kk & 1;
    const unsigned short* Ac = As[cur];
    const unsigned short* Bc = Bs[cur];
    unsigned short* lAn = cur ? lA0 : lA1;
    unsigned short* lBn = cur ? lB0 : lB1;
    int kn = (kk + 1) * 32;                 // next k-step (ushort offset in row)
#if USE_ASYNC_LDS
    if (kk + 1 < NK) {
      __builtin_amdgcn_global_load_async_to_lds_b128(GPTR(Arow + kn), LPTR(lAn), 0, 0);
      __builtin_amdgcn_global_load_async_to_lds_b128(GPTR(Arow + kn + 8), LPTR(lAn + 8), 0, 0);
      __builtin_amdgcn_global_load_async_to_lds_b128(GPTR(Brow + kn), LPTR(lBn), 0, 0);
    }
#else
    v4u a0, a1, b0;
    if (kk + 1 < NK) {
      a0 = *(const v4u*)(Arow + kn);
      a1 = *(const v4u*)(Arow + kn + 8);
      b0 = *(const v4u*)(Brow + kn);
    }
#endif
    // compute on current buffer (overlaps with next-tile fill)
#pragma unroll
    for (int im = 0; im < 2; ++im) {
      v16bf af = load_a_frag_v(Ac + (mW + im * 16 + nlm) * 32 + 8 * lh);
#pragma unroll
      for (int jn = 0; jn < 2; ++jn) {
        v16bf bf = load_b_frag_v(Bc + (nW + jn * 16 + nlm) * 32 + 8 * lh);
        acc[im][jn] = __builtin_amdgcn_wmma_f32_16x16x32_bf16(
            false, af, false, bf, (short)0, acc[im][jn], false, false);
      }
    }
#if USE_ASYNC_LDS
    __builtin_amdgcn_s_wait_asynccnt(0);
#else
    if (kk + 1 < NK) {
      *(v4u*)(lAn) = a0;
      *(v4u*)(lAn + 8) = a1;
      *(v4u*)(lBn) = b0;
    }
#endif
    __syncthreads();
  }

  // epilogue: bias + store in scan-fragment-major layout
  float* preT = prep + (size_t)t * PRE_T_STRIDE;
#pragma unroll
  for (int im = 0; im < 2; ++im)
#pragma unroll
    for (int jn = 0; jn < 2; ++jn) {
      int cb   = n0 + nW + jn * 16;         // global gate-column block base
      int g    = cb >> 8;
      int jt   = (cb >> 4) & 15;
      int tile = (wm * 2 + im) * 16 + jt;   // mt*16 + nt, mt over batch blocks
      float bv = bias[cb + nlm];
      F8 s;
      s.v = acc[im][jn] + bv;
      v4f* d = (v4f*)(preT + ((size_t)tile * 4 + g) * 256 + lane * 8);
      d[0] = s.h[0];
      d[1] = s.h[1];
    }
}

// ---------------- persistent bidirectional LSTM scan ----------------
// grid = 2 blocks (dir 0 fwd, dir 1 bwd), 1024 threads (32 waves).
// Each wave owns 4 (m,j) tiles x 4 gate tiles at identical coordinates;
// c stays in registers, h in LDS (bf16), Whh streams from L2.
__global__ __launch_bounds__(1024)
void lstm_scan_kernel(const float* __restrict__ pre_f,
                      const float* __restrict__ pre_b,
                      const unsigned short* __restrict__ whh_f,
                      const unsigned short* __restrict__ whh_b,
                      float* __restrict__ out) {
  __shared__ unsigned short hls[BB * HHh];            // 64 KB bf16 h-state
  const int dir = blockIdx.x;
  const float* pre = dir ? pre_b : pre_f;
  const unsigned short* Whh = dir ? whh_b : whh_f;

  {
    unsigned int* hz = (unsigned int*)hls;
    for (int i = threadIdx.x; i < BB * HHh / 2; i += blockDim.x) hz[i] = 0u;
  }
  int wave = threadIdx.x >> 5;
  int lane = threadIdx.x & 31;
  int lh = lane >> 4, nl = lane & 15;
  int moff = lh * 8;
  int dcol = dir * HHh;

  // per-lane B pointer: row n = nl, chunk start 8*lh
  const unsigned short* wl = Whh + nl * HHh + 8 * lh;

  int preOff[4][4];   // float offsets into preT
  int wOff[4][4];     // ushort offsets into wl
  int aOff[4];        // ushort offsets into hls (A-frag lane base, minus ks*32)
  int hOff[4];        // ushort offsets into hls for h writeback (e=0)
  int oOff[4];        // float offsets into outT (e=0)
#pragma unroll
  for (int q = 0; q < 4; ++q) {
    int tile = wave * 4 + q;
    int mt = tile >> 4;        // batch block 0..7
    int nt = tile & 15;        // j block 0..15
    int mb = mt * 16, jb = nt * 16;
#pragma unroll
    for (int g = 0; g < 4; ++g) {
      preOff[q][g] = ((tile * 4 + g) * 32 + lane) * 8;
      wOff[q][g]   = (g * HHh + jb) * HHh;
    }
    aOff[q] = (mb + nl) * HHh + 8 * lh;
    hOff[q] = (mb + moff) * HHh + jb + nl;
    oOff[q] = (mb + moff) * (2 * HHh) + dcol + jb + nl;
  }
  v8f cst[4] = {};
  __syncthreads();

  for (int s = 0; s < TT; ++s) {
    int t = dir ? (TT - 1 - s) : s;
    const float* preT = pre + (size_t)t * PRE_T_STRIDE;
    float* outT = out + (size_t)t * (BB * 2 * HHh);
    // seed accumulators with pre-activations: 2 x b128 nontemporal per tile
    v8f acc[4][4];
#pragma unroll
    for (int q = 0; q < 4; ++q)
#pragma unroll
      for (int g = 0; g < 4; ++g) {
        const v4f* pg = (const v4f*)(preT + preOff[q][g]);
        F8 sd;
        sd.h[0] = __builtin_nontemporal_load(pg);
        sd.h[1] = __builtin_nontemporal_load(pg + 1);
        acc[q][g] = sd.v;
      }
    if (s + 1 < TT) {
      int tn = dir ? (TT - 2 - s) : (s + 1);
      __builtin_prefetch(pre + (size_t)tn * PRE_T_STRIDE + preOff[0][0], 0, 1);
    }
    // recurrent GEMM: K=256 -> 8 WMMA k-steps, 16 WMMAs per k-step per wave
    for (int ks = 0; ks < 8; ++ks) {
#pragma unroll
      for (int q = 0; q < 4; ++q) {
        v16bf af = load_a_frag_v(hls + aOff[q] + ks * 32);
#pragma unroll
        for (int g = 0; g < 4; ++g) {
          v16bf bf = load_b_frag_v(wl + wOff[q][g] + ks * 32);
          acc[q][g] = __builtin_amdgcn_wmma_f32_16x16x32_bf16(
              false, af, false, bf, (short)0, acc[q][g], false, false);
        }
      }
    }
    __syncthreads();   // all h reads done before overwriting
    // cell update (gate order i, f, g, o) + write-back
#pragma unroll
    for (int q = 0; q < 4; ++q) {
#pragma unroll
      for (int e = 0; e < 8; ++e) {
        float gi = sig_(acc[q][0][e]);
        float gf = sig_(acc[q][1][e]);
        float gg = tanh_(acc[q][2][e]);
        float go = sig_(acc[q][3][e]);
        float c  = gf * cst[q][e] + gi * gg;
        cst[q][e] = c;
        float h  = go * tanh_(c);
        hls[hOff[q] + e * HHh] = f2bf(h);
        outT[oOff[q] + e * (2 * HHh)] = h;
      }
    }
    __syncthreads();   // h complete before next step's reads
  }
}

// ---------------- emissions: [T*B,512] x [20,512]^T + b ----------------
__global__ __launch_bounds__(256)
void emis_kernel(const float* __restrict__ lo, const float* __restrict__ w,
                 const float* __restrict__ b, float* __restrict__ em) {
  int wid  = blockIdx.x * (blockDim.x >> 5) + (threadIdx.x >> 5);
  int lane = threadIdx.x & 31;
  if (wid >= MM) return;
  const float* xr = lo + (size_t)wid * 512;
  float xv[16];
#pragma unroll
  for (int i = 0; i < 16; ++i) xv[i] = xr[lane + 32 * i];
  for (int k = 0; k < KT; ++k) {
    const float* wr = w + k * 512;
    float s = 0.0f;
#pragma unroll
    for (int i = 0; i < 16; ++i) s += xv[i] * wr[lane + 32 * i];
#pragma unroll
    for (int off = 16; off > 0; off >>= 1) s += __shfl_down(s, off, 32);
    if (lane == 0) em[(size_t)wid * KT + k] = s + b[k];
  }
}

// ---------------- Viterbi decode: one thread per batch element ----------------
__global__ __launch_bounds__(128)
void viterbi_kernel(const float* __restrict__ em, const float* __restrict__ st,
                    const float* __restrict__ et, const float* __restrict__ trans,
                    int* __restrict__ hist, float* __restrict__ tags) {
  __shared__ float tr[KT * KT];
  for (int i = threadIdx.x; i < KT * KT; i += blockDim.x) tr[i] = trans[i];
  __syncthreads();
  int b = threadIdx.x;               // 128 threads
  float score[KT];
#pragma unroll
  for (int k = 0; k < KT; ++k) score[k] = st[k] + em[(size_t)b * KT + k];
  for (int t = 1; t < TT; ++t) {
    float ns[KT];
#pragma unroll
    for (int kn = 0; kn < KT; ++kn) {
      float best = -3.0e38f; int bi = 0;
#pragma unroll
      for (int kp = 0; kp < KT; ++kp) {
        float v = score[kp] + tr[kp * KT + kn];
        if (v > best) { best = v; bi = kp; }
      }
      ns[kn] = best + em[((size_t)t * BB + b) * KT + kn];
      hist[((size_t)(t - 1) * BB + b) * KT + kn] = bi;
    }
#pragma unroll
    for (int k = 0; k < KT; ++k) score[k] = ns[k];
  }
  float best = -3.0e38f; int last = 0;
#pragma unroll
  for (int k = 0; k < KT; ++k) {
    float v = score[k] + et[k];
    if (v > best) { best = v; last = k; }
  }
  tags[(size_t)b * TT + (TT - 1)] = (float)last;
  int tag = last;
  for (int t = TT - 2; t >= 0; --t) {
    tag = hist[((size_t)t * BB + b) * KT + tag];
    tags[(size_t)b * TT + t] = (float)tag;
  }
}

// ---------------- host orchestration ----------------
extern "C" void kernel_launch(void* const* d_in, const int* in_sizes, int n_in,
                              void* d_out, int out_size, void* d_ws, size_t ws_size,
                              hipStream_t stream) {
  (void)in_sizes; (void)n_in; (void)out_size; (void)ws_size;
  const int*   x         = (const int*)d_in[0];
  const float* embedding = (const float*)d_in[2];
  // lstm_params flat order per layer: Wih_f, Whh_f, bih_f, bhh_f, Wih_b, Whh_b, bih_b, bhh_b
  const float* Wih[2][2] = { { (const float*)d_in[3],  (const float*)d_in[7]  },
                             { (const float*)d_in[11], (const float*)d_in[15] } };
  const float* Whh[2][2] = { { (const float*)d_in[4],  (const float*)d_in[8]  },
                             { (const float*)d_in[12], (const float*)d_in[16] } };
  const float* bih[2][2] = { { (const float*)d_in[5],  (const float*)d_in[9]  },
                             { (const float*)d_in[13], (const float*)d_in[17] } };
  const float* bhh[2][2] = { { (const float*)d_in[6],  (const float*)d_in[10] },
                             { (const float*)d_in[14], (const float*)d_in[18] } };
  const float* linear_w  = (const float*)d_in[19];
  const float* linear_b  = (const float*)d_in[20];
  const float* start_tr  = (const float*)d_in[21];
  const float* end_tr    = (const float*)d_in[22];
  const float* trans     = (const float*)d_in[23];
  float* tags = (float*)d_out;

  // carve workspace
  char* p = (char*)d_ws;
  auto carve = [&](size_t bytes) -> void* {
    void* r = (void*)p;
    p += (bytes + 255) & ~(size_t)255;
    return r;
  };
  unsigned short* xin  = (unsigned short*)carve((size_t)MM * 512 * 2);   // layer input bf16
  unsigned short* wihb[2][2];
  wihb[0][0] = (unsigned short*)carve((size_t)GN * KPAD0 * 2);
  wihb[0][1] = (unsigned short*)carve((size_t)GN * KPAD0 * 2);
  wihb[1][0] = (unsigned short*)carve((size_t)GN * 512 * 2);
  wihb[1][1] = (unsigned short*)carve((size_t)GN * 512 * 2);
  unsigned short* whhb[2][2];
  for (int l = 0; l < 2; ++l)
    for (int d = 0; d < 2; ++d) whhb[l][d] = (unsigned short*)carve((size_t)GN * HHh * 2);
  float* biasc[2][2];
  for (int l = 0; l < 2; ++l)
    for (int d = 0; d < 2; ++d) biasc[l][d] = (float*)carve((size_t)GN * 4);
  float* pre_f  = (float*)carve((size_t)TT * PRE_T_STRIDE * 4);   // 256 MB
  float* pre_b  = (float*)carve((size_t)TT * PRE_T_STRIDE * 4);   // 256 MB
  float* outbuf = (float*)carve((size_t)MM * 512 * 4);            // 128 MB
  float* emis   = (float*)carve((size_t)MM * KT * 4);
  int*   hist   = (int*)carve((size_t)(TT - 1) * BB * KT * 4);

  // --- weight conversion to bf16 (+ K padding) & combined biases ---
  for (int l = 0; l < 2; ++l)
    for (int d = 0; d < 2; ++d) {
      int inK  = (l == 0) ? EE : 512;
      int inKp = (l == 0) ? KPAD0 : 512;
      int nW = GN * inKp;
      cvt_pad_bf16_kernel<<<(nW + 255) / 256, 256, 0, stream>>>(Wih[l][d], wihb[l][d], GN, inK, inKp);
      cvt_pad_bf16_kernel<<<(GN * HHh + 255) / 256, 256, 0, stream>>>(Whh[l][d], whhb[l][d], GN, HHh, HHh);
      add_bias_kernel<<<(GN + 255) / 256, 256, 0, stream>>>(bih[l][d], bhh[l][d], biasc[l][d], GN);
    }

  // --- embedding gather -> bf16 padded [MM, 320] ---
  embed_kernel<<<(MM * KPAD0 + 255) / 256, 256, 0, stream>>>(x, embedding, xin);

  // --- layer 0 ---
  {
    dim3 grid(GN / 64, TT);
    gemm_bias_kernel<<<grid, 256, 0, stream>>>(xin, wihb[0][0], biasc[0][0], pre_f, KPAD0);
    gemm_bias_kernel<<<grid, 256, 0, stream>>>(xin, wihb[0][1], biasc[0][1], pre_b, KPAD0);
  }
  lstm_scan_kernel<<<2, 1024, 0, stream>>>(pre_f, pre_b, whhb[0][0], whhb[0][1], outbuf);

  // --- layer 0 output -> bf16 input for layer 1 ---
  cvt_pad_bf16_kernel<<<(MM * 512 + 255) / 256, 256, 0, stream>>>(outbuf, xin, MM, 512, 512);

  // --- layer 1 ---
  {
    dim3 grid(GN / 64, TT);
    gemm_bias_kernel<<<grid, 256, 0, stream>>>(xin, wihb[1][0], biasc[1][0], pre_f, 512);
    gemm_bias_kernel<<<grid, 256, 0, stream>>>(xin, wihb[1][1], biasc[1][1], pre_b, 512);
  }
  lstm_scan_kernel<<<2, 1024, 0, stream>>>(pre_f, pre_b, whhb[1][0], whhb[1][1], outbuf);

  // --- emissions + Viterbi ---
  emis_kernel<<<MM / 8, 256, 0, stream>>>(outbuf, linear_w, linear_b, emis);
  viterbi_kernel<<<1, 128, 0, stream>>>(emis, start_tr, end_tr, trans, hist, tags);
}
